// AddSparseAndLowRankCorrection_83159156785478
// MI455X (gfx1250) — matrix operand order repaired and checked by reading, not compiled, not run on hardware
//
#include <hip/hip_runtime.h>
#include <stdint.h>

#define D_OUT 4096
#define D_IN  4096
#define RANK  64

typedef __attribute__((ext_vector_type(16))) __bf16 v16bf;
typedef __attribute__((ext_vector_type(8)))  float  v8f;

union Frag {
    uint4 q[2];
    v16bf v;
};

// Round-to-nearest-even float -> bf16 (as raw 16-bit pattern)
static __device__ __forceinline__ uint32_t bf16_rne(float f) {
    uint32_t u = __float_as_uint(f);
    return (u + 0x7fffu + ((u >> 16) & 1u)) >> 16;
}

// Split f = hi + lo with hi, lo representable in bf16 (error compensation).
static __device__ __forceinline__ void split2(float f, uint16_t& h, uint16_t& l) {
    uint32_t hb = bf16_rne(f);
    float fh = __uint_as_float(hb << 16);
    uint32_t lb = bf16_rne(f - fh);
    h = (uint16_t)hb;
    l = (uint16_t)lb;
}

// ---------------------------------------------------------------------------
// Phase 1: W_eff = W_inner + A @ B      (W_eff is [D_OUT, D_IN] f32 in d_ws)
// One thread per float4 of output. Within a wave, `o` is uniform -> A loads
// are scalar broadcasts, B loads are coalesced float4.
// ---------------------------------------------------------------------------
__global__ __launch_bounds__(256)
void build_weff_kernel(const float* __restrict__ W,
                       const float* __restrict__ A,
                       const float* __restrict__ Bmat,
                       float* __restrict__ Weff) {
    int gid = blockIdx.x * 256 + threadIdx.x;     // 0 .. D_OUT*(D_IN/4)-1
    int o  = gid >> 10;                           // D_IN/4 = 1024 float4 per row
    int d4 = (gid & 1023) << 2;

    float4 acc = *(const float4*)(W + (size_t)o * D_IN + d4);
    const float* arow = A + (size_t)o * RANK;

#pragma unroll 8
    for (int r = 0; r < RANK; ++r) {
        float a = arow[r];
        float4 b = *(const float4*)(Bmat + (size_t)r * D_IN + d4);
        acc.x += a * b.x;
        acc.y += a * b.y;
        acc.z += a * b.z;
        acc.w += a * b.w;
    }
    *(float4*)(Weff + (size_t)o * D_IN + d4) = acc;
}

// ---------------------------------------------------------------------------
// Phase 2: coalesce the COO sparse matrix into W_eff via f32 atomics (L2).
// sp_indices is int64 [2, NNZ]: rows at [0..NNZ), cols at [NNZ..2*NNZ).
// ---------------------------------------------------------------------------
__global__ __launch_bounds__(256)
void scatter_coo_kernel(const float* __restrict__ vals,
                        const long long* __restrict__ idx,
                        int nnz,
                        float* __restrict__ Weff) {
    int i = blockIdx.x * 256 + threadIdx.x;
    if (i < nnz) {
        int r = (int)idx[i];
        int c = (int)idx[(size_t)nnz + i];
        atomicAdd(Weff + (size_t)r * D_IN + c, vals[i]);
    }
}

// ---------------------------------------------------------------------------
// Phase 3: out[M, D_OUT] = x[M, D_IN] @ W_eff^T  via split-bf16 WMMA.
//
// Block = 256 threads (8 waves, 2x4 wave grid) -> 128x128 output tile.
// Each wave owns a 64x32 sub-tile = 4x2 WMMA tiles, f32 accumulators.
// Per K-step (BK=32): stage f32 x/W tiles, convert to hi/lo bf16 in LDS,
// run 3 WMMAs per tile:  a_hi*b_hi + a_hi*b_lo + a_lo*b_hi  (f32 accum).
// LDS row stride = 40 halves (80 B) so all 16B fragment loads stay aligned
// and bank-conflict pressure is low.
// ---------------------------------------------------------------------------
#define LDK 40   // padded LDS row stride in halves (32 data + 8 pad)

__global__ __launch_bounds__(256)
void gemm_split_bf16_kernel(const float* __restrict__ x,
                            const float* __restrict__ Weff,
                            float* __restrict__ out) {
    __shared__ uint16_t xs_hi[128 * LDK];
    __shared__ uint16_t xs_lo[128 * LDK];
    __shared__ uint16_t ws_hi[128 * LDK];
    __shared__ uint16_t ws_lo[128 * LDK];

    const int tid  = threadIdx.x;
    const int lane = tid & 31;
    const int wid  = tid >> 5;       // 0..7
    const int wm   = wid & 1;        // row half of the 128x128 tile
    const int wn   = wid >> 1;       // 0..3 column slice
    const int lrow = lane & 15;
    const int lhi  = lane >> 4;

    const int n0 = blockIdx.x * 128; // output-channel tile base
    const int m0 = blockIdx.y * 128; // token-row tile base

    v8f acc[4][2];
#pragma unroll
    for (int mt = 0; mt < 4; ++mt)
#pragma unroll
        for (int nt = 0; nt < 2; ++nt)
            acc[mt][nt] = (v8f)0.0f;

#pragma unroll 1
    for (int k0 = 0; k0 < D_IN; k0 += 32) {
        __syncthreads();  // previous iteration's LDS reads done

        // ---- stage + convert: 128 rows x 32 f32 = 1024 float4 per matrix ----
#pragma unroll
        for (int j = 0; j < 4; ++j) {
            int i   = tid + 256 * j;   // float4 index
            int row = i >> 3;          // 8 float4 per row
            int c4  = i & 7;

            float4 xv = *(const float4*)(x + (size_t)(m0 + row) * D_IN + k0 + c4 * 4);
            float4 wv = *(const float4*)(Weff + (size_t)(n0 + row) * D_IN + k0 + c4 * 4);

            uint16_t h[4], l[4];
            split2(xv.x, h[0], l[0]); split2(xv.y, h[1], l[1]);
            split2(xv.z, h[2], l[2]); split2(xv.w, h[3], l[3]);
            uint2 ph, pl;
            ph.x = (uint32_t)h[0] | ((uint32_t)h[1] << 16);
            ph.y = (uint32_t)h[2] | ((uint32_t)h[3] << 16);
            pl.x = (uint32_t)l[0] | ((uint32_t)l[1] << 16);
            pl.y = (uint32_t)l[2] | ((uint32_t)l[3] << 16);
            *(uint2*)(xs_hi + row * LDK + c4 * 4) = ph;
            *(uint2*)(xs_lo + row * LDK + c4 * 4) = pl;

            split2(wv.x, h[0], l[0]); split2(wv.y, h[1], l[1]);
            split2(wv.z, h[2], l[2]); split2(wv.w, h[3], l[3]);
            ph.x = (uint32_t)h[0] | ((uint32_t)h[1] << 16);
            ph.y = (uint32_t)h[2] | ((uint32_t)h[3] << 16);
            pl.x = (uint32_t)l[0] | ((uint32_t)l[1] << 16);
            pl.y = (uint32_t)l[2] | ((uint32_t)l[3] << 16);
            *(uint2*)(ws_hi + row * LDK + c4 * 4) = ph;
            *(uint2*)(ws_lo + row * LDK + c4 * 4) = pl;
        }
        __syncthreads();

        // ---- B fragments: lane holds col N=lrow, K = lhi*16 + 0..15 (contig) ----
        Frag bh[2], bl[2];
#pragma unroll
        for (int nt = 0; nt < 2; ++nt) {
            int n = wn * 32 + nt * 16 + lrow;
            const uint16_t* ph = ws_hi + n * LDK + lhi * 16;
            const uint16_t* pl = ws_lo + n * LDK + lhi * 16;
            bh[nt].q[0] = *(const uint4*)(ph);
            bh[nt].q[1] = *(const uint4*)(ph + 8);
            bl[nt].q[0] = *(const uint4*)(pl);
            bl[nt].q[1] = *(const uint4*)(pl + 8);
        }

        // ---- A fragments + WMMA: lane holds row M=lrow,
        //      K = lhi*16 + 0..15 (contig) ----
#pragma unroll
        for (int mt = 0; mt < 4; ++mt) {
            int m = wm * 64 + mt * 16 + lrow;
            Frag ah, al;
            const uint16_t* pa = xs_hi + m * LDK;
            const uint16_t* pb = xs_lo + m * LDK;
            ah.q[0] = *(const uint4*)(pa + lhi * 16);
            ah.q[1] = *(const uint4*)(pa + lhi * 16 + 8);
            al.q[0] = *(const uint4*)(pb + lhi * 16);
            al.q[1] = *(const uint4*)(pb + lhi * 16 + 8);

#pragma unroll
            for (int nt = 0; nt < 2; ++nt) {
                acc[mt][nt] = __builtin_amdgcn_wmma_f32_16x16x32_bf16(
                    false, ah.v, false, bh[nt].v, (short)0, acc[mt][nt], false, false);
                acc[mt][nt] = __builtin_amdgcn_wmma_f32_16x16x32_bf16(
                    false, ah.v, false, bl[nt].v, (short)0, acc[mt][nt], false, false);
                acc[mt][nt] = __builtin_amdgcn_wmma_f32_16x16x32_bf16(
                    false, al.v, false, bh[nt].v, (short)0, acc[mt][nt], false, false);
            }
        }
    }

    // ---- epilogue: C/D layout -> element j at (row j + 8*lhi, col lrow) ----
#pragma unroll
    for (int mt = 0; mt < 4; ++mt)
#pragma unroll
        for (int nt = 0; nt < 2; ++nt) {
            int col = n0 + wn * 32 + nt * 16 + lrow;
#pragma unroll
            for (int j = 0; j < 8; ++j) {
                int row = m0 + wm * 64 + mt * 16 + j + 8 * lhi;
                out[(size_t)row * D_OUT + col] = acc[mt][nt][j];
            }
        }
}

// ---------------------------------------------------------------------------
// Host-side launcher
// ---------------------------------------------------------------------------
extern "C" void kernel_launch(void* const* d_in, const int* in_sizes, int n_in,
                              void* d_out, int out_size, void* d_ws, size_t ws_size,
                              hipStream_t stream) {
    const float*     x   = (const float*)d_in[0];      // [M, D_IN]
    const float*     W   = (const float*)d_in[1];      // [D_OUT, D_IN]
    const float*     A   = (const float*)d_in[2];      // [D_OUT, RANK]
    const float*     B   = (const float*)d_in[3];      // [RANK, D_IN]
    const float*     sv  = (const float*)d_in[4];      // [NNZ]
    const long long* si  = (const long long*)d_in[5];  // [2, NNZ] int64
    float*           out = (float*)d_out;
    float*           Weff = (float*)d_ws;              // [D_OUT, D_IN] f32

    const size_t weff_bytes = (size_t)D_OUT * D_IN * sizeof(float);
    if (ws_size < weff_bytes) return;  // cannot run without scratch

    int nnz   = in_sizes[4];
    int Mrows = in_sizes[0] / D_IN;    // 8192

    // Phase 1: W_eff = W + A@B
    build_weff_kernel<<<dim3((D_OUT * (D_IN / 4)) / 256), 256, 0, stream>>>(W, A, B, Weff);

    // Phase 2: W_eff += scatter(sp)
    scatter_coo_kernel<<<dim3((nnz + 255) / 256), 256, 0, stream>>>(sv, si, nnz, Weff);

    // Phase 3: out = x @ W_eff^T (split-bf16 WMMA)
    gemm_split_bf16_kernel<<<dim3(D_OUT / 128, Mrows / 128), 256, 0, stream>>>(x, Weff, out);
}
